// SpatialAttention2d_47227460387203
// MI455X (gfx1250) — matrix operand order repaired and checked by reading, not compiled, not run on hardware
//
#include <hip/hip_runtime.h>

#define BB 4
#define CC 32
#define HH 384
#define WW 384
#define KK 16
#define LL 4
#define ROWS 8                   // output rows per conv1 workgroup (wave==row)
#define NPIX (BB*HH*WW)          // 589,824 (batchnorm reduction size)

typedef _Float16     v16h __attribute__((ext_vector_type(16)));
typedef float        v8f  __attribute__((ext_vector_type(8)));
typedef unsigned int v4u  __attribute__((ext_vector_type(4)));

union FragH { v16h h; v4u q[2]; };

__device__ __forceinline__ unsigned short f32_to_f16_bits(float f) {
    union { _Float16 h; unsigned short s; } cv;
    cv.h = (_Float16)f;
    return cv.s;
}

__device__ __forceinline__ int reflect1(int i, int n) {
    i = (i < 0) ? -i : i;            // pad==1 -> single-step reflect
    i = (i >= n) ? (2*n - 2 - i) : i;
    return i;
}

// ---------------------------------------------------------------------------
// K0: zero the atomic stat accumulators (sum[16], sumsq[16], scale[16],
// shift[16], osum, osumsq) so each launch is self-contained.
// ---------------------------------------------------------------------------
__global__ void sa2d_init_stats(float* __restrict__ stat) {
    int i = threadIdx.x;
    if (i < 66) stat[i] = 0.0f;
}

// ---------------------------------------------------------------------------
// K1: conv1 (3x3 reflect, 32->16 ch) as implicit GEMM on WMMA f16 tiles.
// One workgroup = 8 output rows x 128 pixels; stages 10 input rows (1.25x
// read redundancy). Wave w owns output row w: 8 tiles x 9 taps = 72 WMMAs,
// with the 9 B fragments register-resident across all tiles. Fused extras
// while the 32 channels sit in LDS:
//   - per-pixel channel avg/max/min -> attention channels 0..2
//   - per-channel sum/sumsq for batchnorm (global f32 atomics)
// z is stored as f16 with one packed b128 store per lane per tile.
// ---------------------------------------------------------------------------
__global__ __launch_bounds__(256) void sa2d_conv1_wmma(
    const float* __restrict__ x, const float* __restrict__ rw,
    _Float16* __restrict__ z, float* __restrict__ att, float* __restrict__ stat)
{
    __shared__ __align__(16) unsigned short xt[(ROWS + 2) * 130 * 32]; // [row][col][ch]
    __shared__ __align__(16) unsigned short wt[9 * 16 * 32];           // [tap][n][ch]

    const int blk = blockIdx.x;
    const int b   = blk / ((HH / ROWS) * 3);
    const int rem = blk % ((HH / ROWS) * 3);
    const int y0  = (rem / 3) * ROWS;
    const int seg = rem % 3;
    const int px0 = seg * 128;
    const int tid = threadIdx.x;

    // Stage weights: rw is (16,32,3,3); LDS layout tap-major, N, C-contig.
    for (int i = tid; i < 9 * 16 * 32; i += 256) {
        int tap = i >> 9;
        int r   = i & 511;
        int n   = r >> 5;
        int c   = r & 31;
        wt[i] = f32_to_f16_bits(rw[(n * 32 + c) * 9 + tap]);
    }

    // Stage x slab: rows y0-1..y0+ROWS (reflected), cols px0-1..px0+128, 32ch.
    // Global reads coalesced over col; LDS written channel-contiguous.
    for (int s = tid; s < (ROWS + 2) * 130 * 32; s += 256) {
        int row = s / (32 * 130);
        int c   = (s / 130) & 31;
        int col = s % 130;
        int gy  = reflect1(y0 + row - 1, HH);
        int gx  = reflect1(px0 - 1 + col, WW);
        xt[(row * 130 + col) * 32 + c] =
            f32_to_f16_bits(x[((size_t)(b * CC + c) * HH + gy) * WW + gx]);
    }
    __syncthreads();

    // Fused channel avg/max/min over the ROWS*128 interior pixels (att 0..2).
    for (int i = tid; i < ROWS * 128; i += 256) {
        int r = i >> 7;
        int p = i & 127;
        const unsigned short* ps = &xt[((r + 1) * 130 + (p + 1)) * 32];
        float av = 0.f, mx = -3.402823e38f, mn = 3.402823e38f;
        #pragma unroll
        for (int c = 0; c < CC; ++c) {
            union { unsigned short s; _Float16 h; } cv;
            cv.s = ps[c];
            float v = (float)cv.h;
            av += v;
            mx = fmaxf(mx, v);
            mn = fminf(mn, v);
        }
        av *= (1.0f / (float)CC);
        size_t base = ((size_t)(b * 4) * HH + (y0 + r)) * WW + (px0 + p);
        att[base]                       = av;
        att[base + (size_t)HH * WW]     = mx;
        att[base + (size_t)2 * HH * WW] = mn;
    }

    const int wave = tid >> 5;
    const int lane = tid & 31;
    const int half = lane >> 4;        // 0: lanes 0-15, 1: lanes 16-31
    const int mlo  = lane & 15;
    const int r    = wave;             // output row within tile (0..7)
    const int y    = y0 + r;

    // Hoist the 9 B fragments (32x16 f16 each): lane N = mlo; half-wave
    // selects K 0..15 / 16..31 (channel-contiguous in LDS -> 2x ds_load_b128).
    FragH fb[9];
    #pragma unroll
    for (int tap = 0; tap < 9; ++tap) {
        const unsigned short* bp = &wt[(tap * 16 + mlo) * 32 + half * 16];
        fb[tap].q[0] = *(const v4u*)(bp);
        fb[tap].q[1] = *(const v4u*)(bp + 8);
    }

    const int n = mlo;                 // this lane's output channel
    float s = 0.f, s2 = 0.f;
    #pragma unroll 2                   // depth-2 pipeline, bounded VGPR use
    for (int t = 0; t < 8; ++t) {
        const int m0 = t * 16;
        v8f acc = {0.f, 0.f, 0.f, 0.f, 0.f, 0.f, 0.f, 0.f};

        #pragma unroll
        for (int tap = 0; tap < 9; ++tap) {
            const int ky = tap / 3;
            const int kx = tap % 3;
            // A fragment (16x32 f16): lane M = mlo;
            // half 0 -> K 0..15, half 1 -> K 16..31.
            FragH fa;
            const unsigned short* ap =
                &xt[((r + ky) * 130 + (m0 + mlo + kx)) * 32];
            const int cb = half * 16;
            fa.q[0] = *(const v4u*)(ap + cb);
            fa.q[1] = *(const v4u*)(ap + cb + 8);

            acc = __builtin_amdgcn_wmma_f32_16x16x32_f16(
                false, fa.h, false, fb[tap].h, (short)0, acc, false, false);
        }

        // D layout: lane holds out-channel n = mlo, pixels m = j + 8*half
        // -> 8 contiguous pixels; pack to f16, one 16B store (16B-aligned).
        union { v4u u; _Float16 h[8]; } pk;
        #pragma unroll
        for (int j = 0; j < 8; ++j) {
            float v = acc[j];
            pk.h[j] = (_Float16)v;
            s  += v;
            s2 += v * v;
        }
        const int gx0 = px0 + m0 + half * 8;
        *(v4u*)&z[((size_t)(b * KK + n) * HH + y) * WW + gx0] = pk.u;
    }
    atomicAdd(&stat[n], s);
    atomicAdd(&stat[16 + n], s2);
}

// ---------------------------------------------------------------------------
// K2: fold batchnorm 1 into per-channel scale/shift.
// ---------------------------------------------------------------------------
__global__ void sa2d_bn1_fold(const float* __restrict__ rg,
                              const float* __restrict__ rb,
                              float* __restrict__ stat)
{
    int n = threadIdx.x;
    if (n < KK) {
        const float Ninv = 1.0f / (float)NPIX;
        float mean = stat[n] * Ninv;
        float var  = stat[16 + n] * Ninv - mean * mean;
        float inv  = rsqrtf(var + 1e-5f);
        float sc   = rg[n] * inv;
        stat[32 + n] = sc;
        stat[48 + n] = rb[n] - mean * sc;
    }
}

// ---------------------------------------------------------------------------
// K3: per-pixel radius -> 64 bilinear gathers -> contrast (att channel 3).
// Image is 2.4 MB and entirely L2-resident, so the gathers are cache hits.
// ---------------------------------------------------------------------------
__global__ __launch_bounds__(256) void sa2d_contrast(
    const float* __restrict__ image, const _Float16* __restrict__ z,
    const float* __restrict__ stat, float* __restrict__ att)
{
    int p = blockIdx.x * 256 + threadIdx.x;
    if (p >= NPIX) return;
    const int b  = p / (HH * WW);
    const int r  = p % (HH * WW);
    const int y  = r / WW;
    const int xc = r % WW;

    const float* img = image + (size_t)b * HH * WW;
    const float imgc = img[y * WW + xc];

    float ssum = 0.f;
    for (int k = 0; k < KK; ++k) {
        float zz  = (float)z[((size_t)(b * KK + k) * HH + y) * WW + xc];
        float t   = zz * stat[32 + k] + stat[48 + k];
        float sig = 1.0f / (1.0f + __expf(-t));
        float rad = sig * 8.0f + 1.0f;                  // MAX_SCALE, MIN_SCALE
        float th  = 6.283185307179586f * (float)k * (1.0f / 16.0f);
        float dy  = __sinf(th);
        float dx  = __cosf(th);
        #pragma unroll
        for (int l = 0; l < LL; ++l) {
            float rr = rad * (float)(l + 1) * 0.25f;
            float yy = fminf(fmaxf((float)y  + rr * dy, 0.f), (float)(HH - 1));
            float xx = fminf(fmaxf((float)xc + rr * dx, 0.f), (float)(WW - 1));
            int   y0 = min(max((int)floorf(yy), 0), HH - 2);
            int   x0 = min(max((int)floorf(xx), 0), WW - 2);
            float wy = yy - (float)y0;
            float wx = xx - (float)x0;
            const float* row0 = img + (size_t)y0 * WW;
            float v00 = row0[x0],      v01 = row0[x0 + 1];
            float v10 = row0[WW + x0], v11 = row0[WW + x0 + 1];
            ssum += v00 * (1.f - wy) * (1.f - wx) + v01 * (1.f - wy) * wx
                  + v10 * wy * (1.f - wx)         + v11 * wy * wx;
        }
    }
    const float contrast = imgc - ssum * (1.0f / (float)(KK * LL));

    att[((size_t)(b * 4 + 3) * HH + y) * WW + xc] = contrast;
}

// ---------------------------------------------------------------------------
// K4: conv2 (3x3 reflect, 4->1 ch) + block-reduced sum/sumsq for batchnorm 2.
// ---------------------------------------------------------------------------
__global__ __launch_bounds__(256) void sa2d_conv2(
    const float* __restrict__ att, const float* __restrict__ aw,
    float* __restrict__ yraw, float* __restrict__ stat)
{
    __shared__ float w2[36];
    __shared__ float red[256];
    __shared__ float red2[256];
    const int tid = threadIdx.x;
    if (tid < 36) w2[tid] = aw[tid];
    __syncthreads();

    const int p = blockIdx.x * 256 + tid;
    float val = 0.f;
    const bool live = (p < NPIX);
    if (live) {
        const int b  = p / (HH * WW);
        const int r  = p % (HH * WW);
        const int y  = r / WW;
        const int xc = r % WW;
        for (int ch = 0; ch < 4; ++ch) {
            const float* ap = att + (size_t)(b * 4 + ch) * HH * WW;
            #pragma unroll
            for (int ky = 0; ky < 3; ++ky) {
                int gy = reflect1(y + ky - 1, HH);
                #pragma unroll
                for (int kx = 0; kx < 3; ++kx) {
                    int gx = reflect1(xc + kx - 1, WW);
                    val += ap[(size_t)gy * WW + gx] * w2[(ch * 3 + ky) * 3 + kx];
                }
            }
        }
        yraw[p] = val;
    }
    red[tid]  = live ? val : 0.f;
    red2[tid] = live ? val * val : 0.f;
    __syncthreads();
    for (int off = 128; off > 0; off >>= 1) {
        if (tid < off) { red[tid] += red[tid + off]; red2[tid] += red2[tid + off]; }
        __syncthreads();
    }
    if (tid == 0) {
        atomicAdd(&stat[64], red[0]);
        atomicAdd(&stat[65], red2[0]);
    }
}

// ---------------------------------------------------------------------------
// K5: batchnorm 2 + tanh finalize.
// ---------------------------------------------------------------------------
__global__ __launch_bounds__(256) void sa2d_final(
    const float* __restrict__ yraw, const float* __restrict__ stat,
    const float* __restrict__ ag, const float* __restrict__ ab,
    float* __restrict__ out)
{
    int p = blockIdx.x * 256 + threadIdx.x;
    if (p >= NPIX) return;
    const float Ninv = 1.0f / (float)NPIX;
    float mean = stat[64] * Ninv;
    float var  = stat[65] * Ninv - mean * mean;
    float sc   = ag[0] * rsqrtf(var + 1e-5f);
    float sh   = ab[0] - mean * sc;
    out[p] = tanhf(yraw[p] * sc + sh);
}

// ---------------------------------------------------------------------------
// Launcher. Workspace (byte offsets from d_ws):
//   z    : B*16*H*W f16  = 18,874,368 B
//   att  : B*4*H*W  f32  =  9,437,184 B
//   yraw : B*H*W    f32  =  2,359,296 B
//   stat : 66 f32 (sum16, sumsq16, scale16, shift16, osum, osumsq)
// Total ~30.7 MB.
// ---------------------------------------------------------------------------
extern "C" void kernel_launch(void* const* d_in, const int* in_sizes, int n_in,
                              void* d_out, int out_size, void* d_ws, size_t ws_size,
                              hipStream_t stream)
{
    const float* image = (const float*)d_in[0];
    const float* x     = (const float*)d_in[1];
    const float* rw    = (const float*)d_in[2];
    const float* rg    = (const float*)d_in[3];
    const float* rb    = (const float*)d_in[4];
    const float* aw    = (const float*)d_in[5];
    const float* ag    = (const float*)d_in[6];
    const float* ab    = (const float*)d_in[7];
    float* out = (float*)d_out;

    char* base = (char*)d_ws;
    _Float16* z    = (_Float16*)base;
    float*    att  = (float*)(base + (size_t)BB * KK * HH * WW * sizeof(_Float16));
    float*    yraw = (float*)((char*)att + (size_t)BB * 4 * HH * WW * sizeof(float));
    float*    stat = (float*)((char*)yraw + (size_t)NPIX * sizeof(float));

    const int npix_blocks = (NPIX + 255) / 256;

    sa2d_init_stats<<<1, 128, 0, stream>>>(stat);
    sa2d_conv1_wmma<<<BB * (HH / ROWS) * 3, 256, 0, stream>>>(x, rw, z, att, stat);
    sa2d_bn1_fold<<<1, 32, 0, stream>>>(rg, rb, stat);
    sa2d_contrast<<<npix_blocks, 256, 0, stream>>>(image, z, stat, att);
    sa2d_conv2<<<npix_blocks, 256, 0, stream>>>(att, aw, yraw, stat);
    sa2d_final<<<npix_blocks, 256, 0, stream>>>(yraw, stat, ag, ab, out);
}
